// MultiheadAttention_49323404427789
// MI455X (gfx1250) — matrix operand order repaired and checked by reading, not compiled, not run on hardware
//
#include <hip/hip_runtime.h>
#include <hip/hip_bf16.h>

typedef __attribute__((ext_vector_type(16))) __bf16 v16bf;
typedef __attribute__((ext_vector_type(8)))  float  v8f;

constexpr int Bq    = 4;
constexpr int Sq    = 2048;
constexpr int Dm    = 1024;
constexpr int Hh    = 16;
constexpr int HDq   = 64;
constexpr int Mrows = Bq * Sq;   // 8192

__device__ __forceinline__ unsigned short f2bf(float f) {
    unsigned int u = __float_as_uint(f);
    u += 0x7FFFu + ((u >> 16) & 1u);       // round-to-nearest-even
    return (unsigned short)(u >> 16);
}

union AFrag { v16bf v; unsigned int u[8]; };
union BFrag { v16bf v; unsigned int u[8]; unsigned short s[16]; };

// ---------------------------------------------------------------- converters
__global__ void __launch_bounds__(256) f32_to_bf16_kernel(
    const float* __restrict__ in, unsigned short* __restrict__ out, int n) {
    int idx = blockIdx.x * blockDim.x + threadIdx.x;
    if (idx < n) out[idx] = f2bf(in[idx]);
}

// in: [K,N] f32 row-major -> out: [N,K] bf16 (transposed)
__global__ void __launch_bounds__(256) transpose_f32_to_bf16_kernel(
    const float* __restrict__ in, unsigned short* __restrict__ out, int K, int N) {
    int idx = blockIdx.x * blockDim.x + threadIdx.x;
    if (idx < K * N) {
        int k = idx / N, n = idx % N;
        out[(size_t)n * K + k] = f2bf(in[idx]);
    }
}

// V [B*S, D] bf16 -> Vt [B, H, HD, S] bf16 (coalesced writes)
__global__ void __launch_bounds__(256) transpose_v_kernel(
    const unsigned short* __restrict__ Vb, unsigned short* __restrict__ Vt) {
    int idx = blockIdx.x * blockDim.x + threadIdx.x;   // index in Vt layout
    int s  = idx & (Sq - 1);
    int hd = (idx >> 11) & (HDq - 1);
    int h  = (idx >> 17) & (Hh - 1);
    int b  = idx >> 21;
    Vt[idx] = Vb[((size_t)b * Sq + s) * Dm + h * HDq + hd];
}

// ---------------------------------------------------------------- GEMM
// C[M,N] = A[M,K](bf16) @ Bt[N,K](bf16, pre-transposed) (+ bias)
// one wave computes a 32x64 tile: all 6 fragment loads of a k-step are
// batched ahead of the 8 v_wmma_f32_16x16x32_bf16 so the compiler can
// clause them and use partial loadcnt waits.
__global__ void __launch_bounds__(128) gemm_bf16_32x64(
    const unsigned short* __restrict__ A,
    const unsigned short* __restrict__ Bt,
    const float*          __restrict__ bias,   // [N] or nullptr
    float*                __restrict__ outF,   // fp32 out (or nullptr)
    unsigned short*       __restrict__ outB,   // bf16 out (if outF null)
    int M, int N, int K)
{
    const int lane = threadIdx.x & 31;
    const int wid  = blockIdx.x * (blockDim.x >> 5) + (threadIdx.x >> 5);
    const int ntn  = N >> 6;
    const int mt   = wid / ntn;
    const int nt   = wid % ntn;
    const int m0   = mt << 5;
    const int n0   = nt << 6;
    if (m0 >= M) return;

    const int l16 = lane & 15;
    const int hi  = lane >> 4;

    v8f acc[2][4];
    #pragma unroll
    for (int mf = 0; mf < 2; ++mf)
        #pragma unroll
        for (int t = 0; t < 4; ++t)
            #pragma unroll
            for (int j = 0; j < 8; ++j) acc[mf][t][j] = 0.0f;

    const unsigned short* ap0 = A + (size_t)(m0 + l16) * K;
    const unsigned short* ap1 = A + (size_t)(m0 + 16 + l16) * K;
    const unsigned short* bp0 = Bt + (size_t)(n0 + l16) * K + 16 * hi;

    for (int kc = 0; kc < K; kc += 32) {
        // ---- batch all loads of this k-step (12 x b128 after coalescing)
        AFrag a[2];
        #pragma unroll
        for (int j = 0; j < 8; ++j) {
            a[0].u[j] = *(const unsigned int*)(ap0 + kc + 2 * j + 16 * hi);
            a[1].u[j] = *(const unsigned int*)(ap1 + kc + 2 * j + 16 * hi);
        }
        BFrag bm[4];
        #pragma unroll
        for (int t = 0; t < 4; ++t) {
            const unsigned short* bp = bp0 + (size_t)(t * 16) * K + kc;
            #pragma unroll
            for (int j = 0; j < 8; ++j)
                bm[t].u[j] = *(const unsigned int*)(bp + 2 * j);
        }

        // ---- unconditional speculative prefetch, 4 k-steps (256 B) ahead
        __builtin_prefetch(ap0 + kc + 128, 0, 3);
        __builtin_prefetch(ap1 + kc + 128, 0, 3);
        #pragma unroll
        for (int t = 0; t < 4; ++t)
            __builtin_prefetch(bp0 + (size_t)(t * 16) * K + kc + 128, 0, 3);

        // ---- 8 WMMAs, each A-fragment used 4x, each B-fragment 2x
        #pragma unroll
        for (int t = 0; t < 4; ++t) {
            acc[0][t] = __builtin_amdgcn_wmma_f32_16x16x32_bf16(
                false, a[0].v, false, bm[t].v, (short)0, acc[0][t], false, false);
            acc[1][t] = __builtin_amdgcn_wmma_f32_16x16x32_bf16(
                false, a[1].v, false, bm[t].v, (short)0, acc[1][t], false, false);
        }
    }

    #pragma unroll
    for (int mf = 0; mf < 2; ++mf) {
        #pragma unroll
        for (int t = 0; t < 4; ++t) {
            #pragma unroll
            for (int j = 0; j < 8; ++j) {
                int row = m0 + 16 * mf + j + 8 * hi;
                int col = n0 + t * 16 + l16;
                float v = acc[mf][t][j];
                if (bias) v += bias[col];
                if (outF) outF[(size_t)row * N + col] = v;
                else      outB[(size_t)row * N + col] = f2bf(v);
            }
        }
    }
}

// ---------------------------------------------------------------- attention
// Flash-style causal attention: one wave per (b, h, 16-query tile).
// Q/K bf16 in [B*S, 1024]; V pre-transposed bf16 [B, H, HD, S].
__global__ void __launch_bounds__(32) attn_flash_kernel(
    const unsigned short* __restrict__ Q,
    const unsigned short* __restrict__ Kbuf,
    const unsigned short* __restrict__ Vt,
    unsigned short*       __restrict__ ctx)
{
    __shared__ unsigned short plds[16 * 32];   // P tile staging (C->A relayout)

    const int lane = threadIdx.x & 31;
    const int l16  = lane & 15;
    const int hi   = lane >> 4;

    int tile = blockIdx.x;
    const int qtiles = Sq / 16;
    const int qt = tile % qtiles; tile /= qtiles;
    const int h  = tile % Hh;
    const int b  = tile / Hh;
    const int q0 = qt * 16;

    const size_t baseBS = (size_t)b * Sq;
    const unsigned short* Qp  = Q    + baseBS * Dm + h * HDq;
    const unsigned short* Kp  = Kbuf + baseBS * Dm + h * HDq;
    const unsigned short* Vtb = Vt + ((size_t)(b * Hh + h) * HDq) * Sq;

    // Q tile (16x64) as two A fragments (K-chunks of 32), loop invariant.
    AFrag aq[2];
    {
        const unsigned short* qp = Qp + (size_t)(q0 + l16) * Dm;
        #pragma unroll
        for (int c = 0; c < 2; ++c) {
            #pragma unroll
            for (int j = 0; j < 8; ++j)
                aq[c].u[j] = *(const unsigned int*)(qp + c * 32 + 2 * j + 16 * hi);
        }
    }

    float mrow[8], lrow[8];
    v8f   ctxacc[4];
    #pragma unroll
    for (int j = 0; j < 8; ++j) { mrow[j] = -__builtin_inff(); lrow[j] = 0.0f; }
    #pragma unroll
    for (int t = 0; t < 4; ++t)
        #pragma unroll
        for (int j = 0; j < 8; ++j) ctxacc[t][j] = 0.0f;

    // 32 keys per step; last block is the one containing the diagonal.
    for (int kb = 0; kb <= q0 + 15; kb += 32) {
        // ---- batch-load all K fragments for this key block
        BFrag bk[2][2];    // [t][c]: B[k=feature][n=key], contiguous per lane
        #pragma unroll
        for (int t = 0; t < 2; ++t) {
            const unsigned short* kp0 = Kp + (size_t)(kb + t * 16 + l16) * Dm;
            __builtin_prefetch(kp0 + (size_t)32 * Dm, 0, 3);   // next key block
            #pragma unroll
            for (int c = 0; c < 2; ++c) {
                const unsigned short* kp = kp0 + c * 32 + 16 * hi;
                #pragma unroll
                for (int j = 0; j < 8; ++j)
                    bk[t][c].u[j] = *(const unsigned int*)(kp + 2 * j);
            }
        }

        // ---- scores: S = Q @ K^T, two 16x16 C-fragments (keys kb..kb+31)
        v8f sc[2];
        #pragma unroll
        for (int t = 0; t < 2; ++t) {
            #pragma unroll
            for (int j = 0; j < 8; ++j) sc[t][j] = 0.0f;
            #pragma unroll
            for (int c = 0; c < 2; ++c)
                sc[t] = __builtin_amdgcn_wmma_f32_16x16x32_bf16(
                    false, aq[c].v, false, bk[t][c].v, (short)0, sc[t], false, false);
        }

        // ---- online softmax (row m = j + 8*hi spans 16 lanes of a half-wave)
        #pragma unroll
        for (int j = 0; j < 8; ++j) {
            const int qg = q0 + j + 8 * hi;
            #pragma unroll
            for (int t = 0; t < 2; ++t) {
                const int kg = kb + t * 16 + l16;
                float v = sc[t][j] * 0.125f;          // 1/sqrt(64)
                if (kg > qg) v = -__builtin_inff();   // causal mask
                sc[t][j] = v;
            }
            float r = fmaxf(sc[0][j], sc[1][j]);
            #pragma unroll
            for (int off = 1; off < 16; off <<= 1)
                r = fmaxf(r, __shfl_xor(r, off, 16));
            const float mnew  = fmaxf(mrow[j], r);
            const float scale = __expf(mrow[j] - mnew);
            mrow[j]  = mnew;
            lrow[j] *= scale;
            #pragma unroll
            for (int t = 0; t < 4; ++t) ctxacc[t][j] *= scale;

            const float p0 = __expf(sc[0][j] - mnew);
            const float p1 = __expf(sc[1][j] - mnew);
            sc[0][j] = p0; sc[1][j] = p1;
            float rs = p0 + p1;
            #pragma unroll
            for (int off = 1; off < 16; off <<= 1)
                rs += __shfl_xor(rs, off, 16);
            lrow[j] += rs;
        }

        // ---- batch-load all V fragments (contiguous rows of Vt)
        BFrag bv[4];       // B[k=key][n=hd] <- Vt[hd][key]
        #pragma unroll
        for (int t = 0; t < 4; ++t) {
            const unsigned short* vp =
                Vtb + (size_t)(t * 16 + l16) * Sq + kb + 16 * hi;
            #pragma unroll
            for (int j = 0; j < 8; ++j)
                bv[t].u[j] = *(const unsigned int*)(vp + 2 * j);
            __builtin_prefetch(vp + 32, 0, 3);                 // next key block
        }

        // ---- relayout P: C-fragment -> LDS [m][k] -> A-fragment
        #pragma unroll
        for (int t = 0; t < 2; ++t)
            #pragma unroll
            for (int j = 0; j < 8; ++j)
                plds[(j + 8 * hi) * 32 + t * 16 + l16] = f2bf(sc[t][j]);
        asm volatile("s_wait_dscnt 0" ::: "memory");

        AFrag pa;
        {
            const unsigned short* pp = &plds[l16 * 32];
            #pragma unroll
            for (int j = 0; j < 8; ++j)
                pa.u[j] = *(const unsigned int*)(pp + 2 * j + 16 * hi);
        }
        asm volatile("s_wait_dscnt 0" ::: "memory");

        // ---- ctx += P(16x32) @ V(32x64)
        #pragma unroll
        for (int t = 0; t < 4; ++t)
            ctxacc[t] = __builtin_amdgcn_wmma_f32_16x16x32_bf16(
                false, pa.v, false, bv[t].v, (short)0, ctxacc[t], false, false);
    }

    // ---- normalize and store bf16 ctx in [B*S, 1024] layout
    #pragma unroll
    for (int t = 0; t < 4; ++t) {
        #pragma unroll
        for (int j = 0; j < 8; ++j) {
            const int row = q0 + j + 8 * hi;
            const int col = h * HDq + t * 16 + l16;
            const float o = ctxacc[t][j] / lrow[j];
            ctx[(baseBS + row) * Dm + col] = f2bf(o);
        }
    }
}

// ---------------------------------------------------------------- launcher
extern "C" void kernel_launch(void* const* d_in, const int* in_sizes, int n_in,
                              void* d_out, int out_size, void* d_ws, size_t ws_size,
                              hipStream_t stream) {
    const float* x  = (const float*)d_in[0];
    const float* Wq = (const float*)d_in[1];
    const float* Wk = (const float*)d_in[2];
    const float* Wv = (const float*)d_in[3];
    const float* Wo = (const float*)d_in[4];
    const float* bo = (const float*)d_in[5];

    char* ws = (char*)d_ws;
    const size_t MB = 1ull << 20;
    unsigned short* xb   = (unsigned short*)(ws);              // 16 MB (reused as ctx)
    unsigned short* WqT  = (unsigned short*)(ws + 16 * MB);    //  2 MB
    unsigned short* WkT  = (unsigned short*)(ws + 18 * MB);    //  2 MB
    unsigned short* WvT  = (unsigned short*)(ws + 20 * MB);    //  2 MB
    unsigned short* WoT  = (unsigned short*)(ws + 22 * MB);    //  2 MB
    unsigned short* Qb   = (unsigned short*)(ws + 24 * MB);    // 16 MB
    unsigned short* Kb   = (unsigned short*)(ws + 40 * MB);    // 16 MB
    unsigned short* Vb   = (unsigned short*)(ws + 56 * MB);    // 16 MB
    unsigned short* Vt   = (unsigned short*)(ws + 72 * MB);    // 16 MB (total 88 MB)

    // 1) precision conversion (x and transposed weights to bf16)
    const int nX = Mrows * Dm;
    f32_to_bf16_kernel<<<nX / 256, 256, 0, stream>>>(x, xb, nX);
    const int nW = Dm * Dm;
    transpose_f32_to_bf16_kernel<<<nW / 256, 256, 0, stream>>>(Wq, WqT, Dm, Dm);
    transpose_f32_to_bf16_kernel<<<nW / 256, 256, 0, stream>>>(Wk, WkT, Dm, Dm);
    transpose_f32_to_bf16_kernel<<<nW / 256, 256, 0, stream>>>(Wv, WvT, Dm, Dm);
    transpose_f32_to_bf16_kernel<<<nW / 256, 256, 0, stream>>>(Wo, WoT, Dm, Dm);

    // 2) QKV projections (bf16 out)
    const int gemmBlocks = (Mrows / 32) * (Dm / 64) / 4;   // 4 waves per block
    gemm_bf16_32x64<<<gemmBlocks, 128, 0, stream>>>(xb, WqT, nullptr, nullptr, Qb, Mrows, Dm, Dm);
    gemm_bf16_32x64<<<gemmBlocks, 128, 0, stream>>>(xb, WkT, nullptr, nullptr, Kb, Mrows, Dm, Dm);
    gemm_bf16_32x64<<<gemmBlocks, 128, 0, stream>>>(xb, WvT, nullptr, nullptr, Vb, Mrows, Dm, Dm);

    // 2b) transpose V to [B, H, HD, S] so P@V B-fragments load contiguously
    transpose_v_kernel<<<nX / 256, 256, 0, stream>>>(Vb, Vt);

    // 3) fused causal flash attention (ctx written into xb, which is free now)
    attn_flash_kernel<<<Bq * Hh * (Sq / 16), 32, 0, stream>>>(Qb, Kb, Vt, xb);

    // 4) output projection + bias (fp32 out)
    gemm_bf16_32x64<<<gemmBlocks, 128, 0, stream>>>(xb, WoT, bo, (float*)d_out, nullptr, Mrows, Dm, Dm);
}
